// MultiHeadAttention_4312147165789
// MI455X (gfx1250) — compile-verified, matrix operand order checked by reading, not run on hardware
//
#include <hip/hip_runtime.h>

// ---------------------------------------------------------------------------
// Problem constants (match reference)
// ---------------------------------------------------------------------------
#define Bc 16
#define Sc 2048
#define Hc 256
#define NHc 4
#define HDc 64
#define Mrows (Bc * Sc)        // 32768
#define SCALE 0.125f           // 1/sqrt(64)
#define NEGV  (-4294967295.0f) // -(2^32)+1 per reference

typedef __attribute__((ext_vector_type(16))) __bf16          v16bf;
typedef __attribute__((ext_vector_type(16))) unsigned short  v16u;
typedef __attribute__((ext_vector_type(8)))  unsigned short  v8u;
typedef __attribute__((ext_vector_type(8)))  float           v8f;

__device__ __forceinline__ v8f wmma_bf16(v16bf a, v16bf b, v8f c) {
    return __builtin_amdgcn_wmma_f32_16x16x32_bf16(
        /*neg_a=*/false, a, /*neg_b=*/false, b,
        /*c_mod=*/(short)0, c, /*reuse_a=*/false, /*reuse_b=*/false);
}

// CDNA5 16-bit fragment layout: lane L holds, for fragment element i
// (g=i/8, j=i%8), the K index k = g*16 + (L/16)*8 + j.  The 8 elements of
// each half are K-consecutive -> 16 contiguous bytes per half when the
// K dimension is the fast axis in memory.

// A fragment (16x32 MxK) from bf16 row-major [m][k]; two 16B loads per lane.
__device__ __forceinline__ v16bf load_a_bf16(const unsigned short* __restrict__ p, int ld) {
    int lane = threadIdx.x & 31;
    int m = lane & 15, hi = lane >> 4;
    const unsigned short* r = p + (size_t)m * ld + hi * 8;
    v8u lo = *(const v8u*)(r);
    v8u hi8 = *(const v8u*)(r + 16);
    v16u u = __builtin_shufflevector(lo, hi8, 0,1,2,3,4,5,6,7,8,9,10,11,12,13,14,15);
    return __builtin_bit_cast(v16bf, u);
}

// A fragment from f32 row-major [m][k], converted to bf16 (native packed cvt).
__device__ __forceinline__ v16bf load_a_f32(const float* __restrict__ p, int ld) {
    int lane = threadIdx.x & 31;
    int m = lane & 15, hi = lane >> 4;
    const float* r = p + (size_t)m * ld + hi * 8;
    v8f lo = *(const v8f*)(r);        // 32B -> 2x b128
    v8f hi8 = *(const v8f*)(r + 16);
    v16bf a;
#pragma unroll
    for (int i = 0; i < 8; ++i) { a[i] = (__bf16)lo[i]; a[i + 8] = (__bf16)hi8[i]; }
    return a;
}

// B fragment (32x16 KxN) where memory is [n][k] row-major (transposed source):
// B[k][n] = p[n*ld + k]; k fast-varying -> two 16B loads per lane.
__device__ __forceinline__ v16bf load_bT_bf16(const unsigned short* __restrict__ p, int ld) {
    int lane = threadIdx.x & 31;
    int n = lane & 15, hi = lane >> 4;
    const unsigned short* r = p + (size_t)n * ld + hi * 8;
    v8u lo = *(const v8u*)(r);
    v8u hi8 = *(const v8u*)(r + 16);
    v16u u = __builtin_shufflevector(lo, hi8, 0,1,2,3,4,5,6,7,8,9,10,11,12,13,14,15);
    return __builtin_bit_cast(v16bf, u);
}

// 16-lane-half row reductions (C layout: row spread across a 16-lane half)
__device__ __forceinline__ float row_max16(float v) {
    v = fmaxf(v, __shfl_xor(v, 1, 32));
    v = fmaxf(v, __shfl_xor(v, 2, 32));
    v = fmaxf(v, __shfl_xor(v, 4, 32));
    v = fmaxf(v, __shfl_xor(v, 8, 32));
    return v;
}
__device__ __forceinline__ float row_sum16(float v) {
    v += __shfl_xor(v, 1, 32);
    v += __shfl_xor(v, 2, 32);
    v += __shfl_xor(v, 4, 32);
    v += __shfl_xor(v, 8, 32);
    return v;
}

// ---------------------------------------------------------------------------
// Pass 1: W[256,256] f32 -> fragment-swizzled bf16.
// Layout: Wswz[((kt*16 + nt)*32 + lane)*16 + i], so a GEMM B-fragment is one
// contiguous 32B load per lane.  B[k][n] = W[n][k].
// ---------------------------------------------------------------------------
__global__ void __launch_bounds__(256)
wswz_kernel(const float* __restrict__ W, unsigned short* __restrict__ Wswz) {
    int idx = blockIdx.x * blockDim.x + threadIdx.x;   // 65536 total
    int i    = idx & 15;
    int lane = (idx >> 4) & 31;
    int frag = idx >> 9;           // 0..127
    int kt = frag >> 4, nt = frag & 15;
    int hi = lane >> 4, nl = lane & 15;
    int g = i >> 3, j = i & 7;
    int k = kt * 32 + g * 16 + hi * 8 + j;
    int n = nt * 16 + nl;
    __bf16 b = (__bf16)W[(size_t)n * Hc + k];
    Wswz[idx] = __builtin_bit_cast(unsigned short, b);
}

// ---------------------------------------------------------------------------
// Pass 2: projection GEMM  out = X @ W^T + bias (+ e0+e1+e2), stored bf16 in
// head-major layout:  out[((h*B + b)*S + s)*HD + d],  h = n/HD, d = n%HD.
// 128 threads = 4 waves; each wave computes 16(M) x 256(full N), so every
// f32 activation tile is read from HBM exactly once (A-fragment reused 16x).
// __launch_bounds__(128,1) -> allocator may use the full per-wave VGPR budget
// (CDNA5: up to 1024 VGPRs/wave) so the 16 f32 accumulators stay in registers.
// ---------------------------------------------------------------------------
__global__ void __launch_bounds__(128, 1)
proj_gemm_kernel(const float* __restrict__ X,
                 const unsigned short* __restrict__ Wswz,
                 const float* __restrict__ bias,
                 const float* __restrict__ e0,
                 const float* __restrict__ e1,
                 const float* __restrict__ e2,
                 unsigned short* __restrict__ out,
                 int hasExtras) {
    int wave = threadIdx.x >> 5;
    int lane = threadIdx.x & 31;
    int m0 = blockIdx.x * 64 + wave * 16;
    const v16u* Wf = (const v16u*)Wswz;

    v8f acc[16] = {};
#pragma unroll
    for (int k0 = 0; k0 < Hc; k0 += 32) {
        v16bf a = load_a_f32(X + (size_t)m0 * Hc + k0, Hc);
        int kt = k0 >> 5;
#pragma unroll
        for (int t = 0; t < 16; ++t) {
            v16u u = Wf[((size_t)kt * 16 + t) * 32 + lane];  // contiguous 32B load
            acc[t] = wmma_bf16(a, __builtin_bit_cast(v16bf, u), acc[t]);
        }
    }

    int nl = lane & 15, hi = lane >> 4;
#pragma unroll
    for (int t = 0; t < 16; ++t) {
        int n = t * 16 + nl;
        float bs = bias[n];
        int hh = n >> 6, d = n & 63;
#pragma unroll
        for (int r = 0; r < 8; ++r) {
            int m = m0 + r + 8 * hi;
            float v = acc[t][r] + bs;
            if (hasExtras) {
                size_t idx = (size_t)m * Hc + n;
                v += e0[idx] + e1[idx] + e2[idx];
            }
            int bb = m >> 11, ss = m & (Sc - 1);
            __bf16 bv = (__bf16)v;
            out[(((size_t)(hh * Bc + bb)) * Sc + ss) * HDc + d] =
                __builtin_bit_cast(unsigned short, bv);
        }
    }
}

// ---------------------------------------------------------------------------
// Pass 2b: per-head transpose  Vrow[hb][s][d] -> Vt[hb][d][s] (bf16),
// LDS-tiled so both global read and write are 16B-contiguous per lane.
// ---------------------------------------------------------------------------
__global__ void __launch_bounds__(256)
vtrans_kernel(const unsigned short* __restrict__ Vrow,
              unsigned short* __restrict__ Vt) {
    __shared__ unsigned short tile[64][72];   // pad for bank spread
    int hb = blockIdx.x >> 5;                 // 64 heads*batches
    int st = blockIdx.x & 31;                 // 32 s-tiles of 64
    int s0 = st * 64;
    const unsigned short* src = Vrow + ((size_t)hb * Sc + s0) * HDc;
    unsigned short* dst = Vt + (size_t)hb * HDc * Sc;
    int tid = threadIdx.x;                    // 256 threads

#pragma unroll
    for (int p = 0; p < 2; ++p) {
        int off = p * 2048 + tid * 8;
        int sl = off >> 6, d0 = off & 63;
        v8u v = *(const v8u*)(src + (size_t)sl * HDc + d0);
#pragma unroll
        for (int i = 0; i < 8; ++i) tile[sl][d0 + i] = v[i];
    }
    __syncthreads();
#pragma unroll
    for (int p = 0; p < 2; ++p) {
        int off = p * 2048 + tid * 8;
        int dl = off >> 6, sl0 = off & 63;
        v8u v;
#pragma unroll
        for (int i = 0; i < 8; ++i) v[i] = tile[sl0 + i][dl];
        *(v8u*)(dst + (size_t)dl * Sc + s0 + sl0) = v;
    }
}

// ---------------------------------------------------------------------------
// Pass 3: causal flash attention, one wave per 16-row query tile.
// Qh/Kh: bf16 head-major [(h*B+b)*S + s][d].  Vt: bf16 [(h*B+b)*HD + d][s].
// ---------------------------------------------------------------------------
__global__ void __launch_bounds__(128, 1)
flash_attn_kernel(const unsigned short* __restrict__ Qh,
                  const unsigned short* __restrict__ Kh,
                  const unsigned short* __restrict__ Vt,
                  float* __restrict__ out) {
    __shared__ unsigned short pbuf[4][16 * 32];   // per-wave P tile (bf16)

    int wave = threadIdx.x >> 5;
    int lane = threadIdx.x & 31;
    int nl = lane & 15, hi = lane >> 4;

    const int tilesPerHB = Sc / 64;                 // 32
    int hb = blockIdx.x / tilesPerHB;               // h*B + b
    int qc = blockIdx.x % tilesPerHB;
    int q0 = qc * 64 + wave * 16;

    const unsigned short* Qp = Qh + (size_t)hb * Sc * HDc;
    const unsigned short* Kp = Kh + (size_t)hb * Sc * HDc;
    const unsigned short* Vp = Vt + (size_t)hb * HDc * Sc;
    unsigned short* pb = &pbuf[wave][0];

    // Q fragments (K-dim 64 -> two 16x32 A frags), loaded once
    v16bf aq0 = load_a_bf16(Qp + (size_t)q0 * HDc + 0, HDc);
    v16bf aq1 = load_a_bf16(Qp + (size_t)q0 * HDc + 32, HDc);

    float mrow[8], lrow[8];
    v8f o[4] = {};
#pragma unroll
    for (int r = 0; r < 8; ++r) { mrow[r] = -__builtin_inff(); lrow[r] = 0.0f; }

    int jend = q0 + 16;                             // need columns < q0+16
    for (int j = 0; j < jend; j += 32) {
        // prefetch next K/V block into cache (gfx1250 global_prefetch)
        if (j + 32 < jend) {
            __builtin_prefetch(Kp + (size_t)(j + 32) * HDc, 0, 3);
            __builtin_prefetch(Vp + (size_t)(j + 32), 0, 3);
        }

        // ---- scores for 32 key positions: two 16x16 C tiles ----
        v16bf bk0 = load_bT_bf16(Kp + (size_t)j * HDc + 0, HDc);
        v16bf bk1 = load_bT_bf16(Kp + (size_t)j * HDc + 32, HDc);
        v8f c0 = {};
        c0 = wmma_bf16(aq0, bk0, c0);
        c0 = wmma_bf16(aq1, bk1, c0);

        v16bf bk2 = load_bT_bf16(Kp + (size_t)(j + 16) * HDc + 0, HDc);
        v16bf bk3 = load_bT_bf16(Kp + (size_t)(j + 16) * HDc + 32, HDc);
        v8f c1 = {};
        c1 = wmma_bf16(aq0, bk2, c1);
        c1 = wmma_bf16(aq1, bk3, c1);

        // ---- scale, causal mask, online softmax update, P -> LDS ----
#pragma unroll
        for (int r = 0; r < 8; ++r) {
            int row = q0 + r + 8 * hi;
            float s0 = c0[r] * SCALE;
            float s1 = c1[r] * SCALE;
            if (j + nl > row)      s0 = NEGV;
            if (j + 16 + nl > row) s1 = NEGV;
            float bm = row_max16(fmaxf(s0, s1));
            float newm = fmaxf(mrow[r], bm);
            float alpha = __expf(mrow[r] - newm);
            mrow[r] = newm;
            float p0 = __expf(s0 - newm);
            float p1 = __expf(s1 - newm);
            lrow[r] = lrow[r] * alpha + row_sum16(p0 + p1);
#pragma unroll
            for (int t = 0; t < 4; ++t) o[t][r] *= alpha;
            int m = r + 8 * hi;
            __bf16 b0 = (__bf16)p0, b1 = (__bf16)p1;
            pb[m * 32 + nl]      = __builtin_bit_cast(unsigned short, b0);
            pb[m * 32 + 16 + nl] = __builtin_bit_cast(unsigned short, b1);
        }

        // ---- O += P(16x32) @ Vsum(32x64) ----
        v16bf ap = load_a_bf16(pb, 32);   // LDS re-read in A layout (ds_load_b128)
#pragma unroll
        for (int t = 0; t < 4; ++t) {
            // B[k][n] = Vt[(t*16+n)][j+k]  -> contiguous per lane
            v16bf bv = load_bT_bf16(Vp + (size_t)(t * 16) * Sc + j, Sc);
            o[t] = wmma_bf16(ap, bv, o[t]);
        }
    }

    // ---- normalize and store f32 output [B,S,H], h = head*HD + d ----
    int head = hb / Bc, bb = hb % Bc;
#pragma unroll
    for (int t = 0; t < 4; ++t) {
#pragma unroll
        for (int r = 0; r < 8; ++r) {
            int row = q0 + r + 8 * hi;
            int d = t * 16 + nl;
            out[((size_t)bb * Sc + row) * Hc + head * HDc + d] = o[t][r] / lrow[r];
        }
    }
}

// ---------------------------------------------------------------------------
// Host-side launcher
// ---------------------------------------------------------------------------
extern "C" void kernel_launch(void* const* d_in, const int* in_sizes, int n_in,
                              void* d_out, int out_size, void* d_ws, size_t ws_size,
                              hipStream_t stream) {
    const float* queries  = (const float*)d_in[0];
    const float* keys     = (const float*)d_in[1];
    const float* neigh    = (const float*)d_in[2];
    const float* neighAt  = (const float*)d_in[3];
    const float* absK     = (const float*)d_in[4];
    const float* absV     = (const float*)d_in[5];
    // d_in[6] = attn_mask (causal, known statically -> unused)
    const float* Qw = (const float*)d_in[7];
    const float* Qb = (const float*)d_in[8];
    const float* Kw = (const float*)d_in[9];
    const float* Kb = (const float*)d_in[10];
    const float* Vw = (const float*)d_in[11];
    const float* Vb = (const float*)d_in[12];
    float* out = (float*)d_out;

    // workspace carving (bf16 buffers)
    char* ws = (char*)d_ws;
    const size_t wBytes = (size_t)Hc * Hc * sizeof(unsigned short);    // 128 KB
    const size_t tBytes = (size_t)Mrows * Hc * sizeof(unsigned short); // 16 MB
    unsigned short* Qwz  = (unsigned short*)(ws);
    unsigned short* Kwz  = (unsigned short*)(ws + wBytes);
    unsigned short* Vwz  = (unsigned short*)(ws + 2 * wBytes);
    unsigned short* Qhm  = (unsigned short*)(ws + 3 * wBytes);
    unsigned short* Ksum = (unsigned short*)(ws + 3 * wBytes + tBytes);
    unsigned short* Vrow = (unsigned short*)(ws + 3 * wBytes + 2 * tBytes);
    unsigned short* Vt   = (unsigned short*)(ws + 3 * wBytes + 3 * tBytes);

    // Pass 1: weight convert + fragment swizzle
    {
        dim3 grid(Hc * Hc / 256), block(256);
        wswz_kernel<<<grid, block, 0, stream>>>(Qw, Qwz);
        wswz_kernel<<<grid, block, 0, stream>>>(Kw, Kwz);
        wswz_kernel<<<grid, block, 0, stream>>>(Vw, Vwz);
    }

    // Pass 2: projections (Q plain; Ksum/Vsum with fused additive tensors)
    {
        dim3 grid(Mrows / 64), block(128);
        proj_gemm_kernel<<<grid, block, 0, stream>>>(queries, Qwz, Qb,
                                                     nullptr, nullptr, nullptr, Qhm, 0);
        proj_gemm_kernel<<<grid, block, 0, stream>>>(keys, Kwz, Kb,
                                                     neigh, neighAt, absK, Ksum, 1);
        proj_gemm_kernel<<<grid, block, 0, stream>>>(keys, Vwz, Vb,
                                                     neigh, neighAt, absV, Vrow, 1);
    }

    // Pass 2b: V transpose per head (for contiguous B-fragments in attention)
    {
        dim3 grid(NHc * Bc * (Sc / 64)), block(256);
        vtrans_kernel<<<grid, block, 0, stream>>>(Vrow, Vt);
    }

    // Pass 3: causal flash attention
    {
        dim3 grid(NHc * Bc * (Sc / 64)), block(128);
        flash_attn_kernel<<<grid, block, 0, stream>>>(Qhm, Ksum, Vt, out);
    }
}
